// FeedForward_70944269796048
// MI455X (gfx1250) — 8-bit matrix operand order repaired and checked by reading, compiled, not run on hardware
//
#include <hip/hip_runtime.h>
#include <stdint.h>

typedef int v8i __attribute__((ext_vector_type(8)));

#define EPS      1e-5f
#define THREADS  256
#define TM       256
#define TN       64
#define TK       64
#define LDB      80

__device__ __forceinline__ void async_load_b128(unsigned lds, const int8_t* gp) {
    asm volatile("global_load_async_to_lds_b128 %0, %1, off"
                 :: "v"(lds), "v"((unsigned long long)(uintptr_t)gp)
                 : "memory");
}
__device__ __forceinline__ void wait_async0() {
    asm volatile("s_wait_asynccnt 0x0" ::: "memory");
}

__device__ __forceinline__ void stage_A(unsigned lbase, const int8_t* __restrict__ g,
                                        int K, int m0, int k0, int tid) {
    int row = tid >> 2, seg = (tid & 3) << 4;
#pragma unroll
    for (int i = 0; i < 4; ++i) {
        async_load_b128(lbase + (row + 64 * i) * LDB + seg,
                        g + (long long)(m0 + row + 64 * i) * K + k0 + seg);
    }
}
__device__ __forceinline__ void stage_B(unsigned lbase, const int8_t* __restrict__ g,
                                        int K, int n0, int k0, int tid) {
    int col = tid >> 2, seg = (tid & 3) << 4;
    async_load_b128(lbase + col * LDB + seg,
                    g + (long long)(n0 + col) * K + k0 + seg);
}

__device__ __forceinline__ v8i frag_A(const char* p) {
    v8i a;
    ((int2*)&a)[0] = *(const int2*)(p);
    ((int2*)&a)[1] = *(const int2*)(p + 16);
    ((int2*)&a)[2] = *(const int2*)(p + 32);
    ((int2*)&a)[3] = *(const int2*)(p + 48);
    return a;
}
__device__ __forceinline__ v8i frag_B(const char* p) {
    v8i b;
    ((int2*)&b)[0] = *(const int2*)(p);
    ((int2*)&b)[1] = *(const int2*)(p + 16);
    ((int2*)&b)[2] = *(const int2*)(p + 32);
    ((int2*)&b)[3] = *(const int2*)(p + 48);
    return b;
}

__global__ __launch_bounds__(THREADS)
void kabs_partial(const float* __restrict__ w, long long n, float* __restrict__ partial) {
    __shared__ float sm[THREADS];
    float s = 0.f;
    long long stride = (long long)gridDim.x * THREADS;
    for (long long i = (long long)blockIdx.x * THREADS + threadIdx.x; i < n; i += stride)
        s += fabsf(w[i]);
    sm[threadIdx.x] = s;
    __syncthreads();
    for (int o = THREADS / 2; o > 0; o >>= 1) {
        if (threadIdx.x < o) sm[threadIdx.x] += sm[threadIdx.x + o];
        __syncthreads();
    }
    if (threadIdx.x == 0) partial[blockIdx.x] = sm[0];
}

__global__ __launch_bounds__(THREADS)
void kscale_finalize(const float* __restrict__ partial, float* __restrict__ scale,
                     float* __restrict__ descale, float invn, int nb) {
    __shared__ float sm[THREADS];
    const float* p = partial + blockIdx.x * nb;
    float s = 0.f;
    for (int i = threadIdx.x; i < nb; i += THREADS) s += p[i];
    sm[threadIdx.x] = s;
    __syncthreads();
    for (int o = THREADS / 2; o > 0; o >>= 1) {
        if (threadIdx.x < o) sm[threadIdx.x] += sm[threadIdx.x + o];
        __syncthreads();
    }
    if (threadIdx.x == 0) {
        float mean = sm[0] * invn;
        float m    = fmaxf(mean, EPS);
        scale[blockIdx.x]   = 1.f / m;
        descale[blockIdx.x] = m;
    }
}

__global__ __launch_bounds__(THREADS)
void kquant_w(const float* __restrict__ w, int8_t* __restrict__ wq, long long n4,
              const float* __restrict__ scale, int idx) {
    float s = scale[idx];
    long long stride = (long long)gridDim.x * THREADS;
    for (long long i = (long long)blockIdx.x * THREADS + threadIdx.x; i < n4; i += stride) {
        float4 v = ((const float4*)w)[i];
        char4 q;
        q.x = (int8_t)fminf(1.f, fmaxf(-1.f, rintf(v.x * s)));
        q.y = (int8_t)fminf(1.f, fmaxf(-1.f, rintf(v.y * s)));
        q.z = (int8_t)fminf(1.f, fmaxf(-1.f, rintf(v.z * s)));
        q.w = (int8_t)fminf(1.f, fmaxf(-1.f, rintf(v.w * s)));
        ((char4*)wq)[i] = q;
    }
}

__global__ __launch_bounds__(THREADS)
void kact_quant(const float* __restrict__ x, int8_t* __restrict__ xq,
                float* __restrict__ descale, int D) {
    __shared__ float sm[THREADS];
    long long row = blockIdx.x;
    const float* xr = x + row * D;
    float m = 0.f;
    for (int i = threadIdx.x; i < D; i += THREADS) m = fmaxf(m, fabsf(xr[i]));
    sm[threadIdx.x] = m;
    __syncthreads();
    for (int o = THREADS / 2; o > 0; o >>= 1) {
        if (threadIdx.x < o) sm[threadIdx.x] = fmaxf(sm[threadIdx.x], sm[threadIdx.x + o]);
        __syncthreads();
    }
    float amax  = fmaxf(sm[0], EPS);
    float scale = 127.f / amax;
    if (threadIdx.x == 0) descale[row] = amax * (1.f / 127.f);
    int8_t* qr = xq + row * D;
    for (int i = threadIdx.x; i < D; i += THREADS) {
        float q = rintf(xr[i] * scale);
        q = fminf(127.f, fmaxf(-128.f, q));
        qr[i] = (int8_t)q;
    }
}

__global__ __launch_bounds__(THREADS)
void kgemm_gate_up(const int8_t* __restrict__ xq, const float* __restrict__ xds,
                   const int8_t* __restrict__ wqg, const int8_t* __restrict__ wqu,
                   const float* __restrict__ wdesc, float* __restrict__ h,
                   int N, int K) {
    __shared__ __align__(16) char lA [2][TM * LDB];
    __shared__ __align__(16) char lBg[2][TN * LDB];
    __shared__ __align__(16) char lBu[2][TN * LDB];

    const int tid  = threadIdx.x;
    const int wv   = tid >> 5;
    const int lane = tid & 31;
    const int m0   = blockIdx.x * TM;
    const int n0   = blockIdx.y * TN;

    const unsigned aL [2] = {(unsigned)(uintptr_t)&lA [0][0], (unsigned)(uintptr_t)&lA [1][0]};
    const unsigned bgL[2] = {(unsigned)(uintptr_t)&lBg[0][0], (unsigned)(uintptr_t)&lBg[1][0]};
    const unsigned buL[2] = {(unsigned)(uintptr_t)&lBu[0][0], (unsigned)(uintptr_t)&lBu[1][0]};

    const v8i vz = {0, 0, 0, 0, 0, 0, 0, 0};
    v8i accg[2][4] = {{vz, vz, vz, vz}, {vz, vz, vz, vz}};
    v8i accu[2][4] = {{vz, vz, vz, vz}, {vz, vz, vz, vz}};

    const int aofs0 = (32 * wv + (lane & 15)) * LDB + ((lane >> 4) << 3);
    const int aofs1 = aofs0 + 16 * LDB;
    const int bofs  = (lane & 15) * LDB + ((lane >> 4) << 3);

    stage_A(aL[0],  xq,  K, m0, 0, tid);
    stage_B(bgL[0], wqg, K, n0, 0, tid);
    stage_B(buL[0], wqu, K, n0, 0, tid);
    wait_async0();
    __syncthreads();

    int buf = 0;
    for (int k0 = 0; k0 < K; k0 += TK, buf ^= 1) {
        if (k0 + TK < K) {
            stage_A(aL[buf ^ 1],  xq,  K, m0, k0 + TK, tid);
            stage_B(bgL[buf ^ 1], wqg, K, n0, k0 + TK, tid);
            stage_B(buL[buf ^ 1], wqu, K, n0, k0 + TK, tid);
        }
        const v8i a0 = frag_A(&lA[buf][0] + aofs0);
        const v8i a1 = frag_A(&lA[buf][0] + aofs1);
#pragma unroll
        for (int c = 0; c < 4; ++c) {
            const v8i bg = frag_B(&lBg[buf][0] + c * 16 * LDB + bofs);
            accg[0][c] = __builtin_amdgcn_wmma_i32_16x16x64_iu8(true, a0, true, bg,
                                                                accg[0][c], false, false);
            accg[1][c] = __builtin_amdgcn_wmma_i32_16x16x64_iu8(true, a1, true, bg,
                                                                accg[1][c], false, false);
            const v8i bu = frag_B(&lBu[buf][0] + c * 16 * LDB + bofs);
            accu[0][c] = __builtin_amdgcn_wmma_i32_16x16x64_iu8(true, a0, true, bu,
                                                                accu[0][c], false, false);
            accu[1][c] = __builtin_amdgcn_wmma_i32_16x16x64_iu8(true, a1, true, bu,
                                                                accu[1][c], false, false);
        }
        wait_async0();
        __syncthreads();
    }

    const float dwg = wdesc[0], dwu = wdesc[1];
    const int nbase = n0 + (lane & 15);
#pragma unroll
    for (int s = 0; s < 2; ++s) {
        const int mbase = m0 + 32 * wv + 16 * s + ((lane >> 4) << 3);
        float dx[8];
#pragma unroll
        for (int r = 0; r < 8; ++r) dx[r] = xds[mbase + r];
#pragma unroll
        for (int c = 0; c < 4; ++c) {
#pragma unroll
            for (int r = 0; r < 8; ++r) {
                float g  = (float)accg[s][c][r] * dx[r] * dwg;
                float u  = (float)accu[s][c][r] * dx[r] * dwu;
                float sg = g * (1.f / (1.f + __expf(-g)));
                h[(long long)(mbase + r) * N + nbase + 16 * c] = sg * u;
            }
        }
    }
}

__global__ __launch_bounds__(THREADS)
void kgemm_down(const int8_t* __restrict__ hq, const float* __restrict__ hds,
                const int8_t* __restrict__ wqd, const float* __restrict__ wdesc,
                float* __restrict__ out, int N, int K) {
    __shared__ __align__(16) char lA[2][TM * LDB];
    __shared__ __align__(16) char lB[2][TN * LDB];

    const int tid  = threadIdx.x;
    const int wv   = tid >> 5;
    const int lane = tid & 31;
    const int m0   = blockIdx.x * TM;
    const int n0   = blockIdx.y * TN;

    const unsigned aL[2] = {(unsigned)(uintptr_t)&lA[0][0], (unsigned)(uintptr_t)&lA[1][0]};
    const unsigned bL[2] = {(unsigned)(uintptr_t)&lB[0][0], (unsigned)(uintptr_t)&lB[1][0]};

    const v8i vz = {0, 0, 0, 0, 0, 0, 0, 0};
    v8i acc[2][4] = {{vz, vz, vz, vz}, {vz, vz, vz, vz}};

    const int aofs0 = (32 * wv + (lane & 15)) * LDB + ((lane >> 4) << 3);
    const int aofs1 = aofs0 + 16 * LDB;
    const int bofs  = (lane & 15) * LDB + ((lane >> 4) << 3);

    stage_A(aL[0], hq,  K, m0, 0, tid);
    stage_B(bL[0], wqd, K, n0, 0, tid);
    wait_async0();
    __syncthreads();

    int buf = 0;
    for (int k0 = 0; k0 < K; k0 += TK, buf ^= 1) {
        if (k0 + TK < K) {
            stage_A(aL[buf ^ 1], hq,  K, m0, k0 + TK, tid);
            stage_B(bL[buf ^ 1], wqd, K, n0, k0 + TK, tid);
        }
        const v8i a0 = frag_A(&lA[buf][0] + aofs0);
        const v8i a1 = frag_A(&lA[buf][0] + aofs1);
#pragma unroll
        for (int c = 0; c < 4; ++c) {
            const v8i b = frag_B(&lB[buf][0] + c * 16 * LDB + bofs);
            acc[0][c] = __builtin_amdgcn_wmma_i32_16x16x64_iu8(true, a0, true, b,
                                                               acc[0][c], false, false);
            acc[1][c] = __builtin_amdgcn_wmma_i32_16x16x64_iu8(true, a1, true, b,
                                                               acc[1][c], false, false);
        }
        wait_async0();
        __syncthreads();
    }

    const float dwd = wdesc[2];
    const int nbase = n0 + (lane & 15);
#pragma unroll
    for (int s = 0; s < 2; ++s) {
        const int mbase = m0 + 32 * wv + 16 * s + ((lane >> 4) << 3);
        float dh[8];
#pragma unroll
        for (int r = 0; r < 8; ++r) dh[r] = hds[mbase + r];
#pragma unroll
        for (int c = 0; c < 4; ++c) {
#pragma unroll
            for (int r = 0; r < 8; ++r) {
                out[(long long)(mbase + r) * N + nbase + 16 * c] =
                    (float)acc[s][c][r] * dh[r] * dwd;
            }
        }
    }
}

extern "C" void kernel_launch(void* const* d_in, const int* in_sizes, int n_in,
                              void* d_out, int out_size, void* d_ws, size_t ws_size,
                              hipStream_t stream) {
    (void)in_sizes; (void)n_in; (void)out_size; (void)ws_size;
    const float* x  = (const float*)d_in[0];
    const float* wg = (const float*)d_in[1];
    const float* wu = (const float*)d_in[2];
    const float* wd = (const float*)d_in[3];
    float* out = (float*)d_out;

    const int B = 4, S = 2048, D = 2048, I = 6464;
    const int M = B * S;
    const long long NW = (long long)I * D;
    const int NB = 512;

    char*  ws       = (char*)d_ws;
    float* scale    = (float*)ws;
    float* descale  = scale + 4;
    float* partials = scale + 16;
    size_t cur = (((16 + 3 * (size_t)NB) * sizeof(float)) + 255) & ~(size_t)255;
    int8_t* xq  = (int8_t*)(ws + cur); cur += (size_t)M * D;
    float*  xds = (float*) (ws + cur); cur += (size_t)M * sizeof(float);
    int8_t* wqg = (int8_t*)(ws + cur); cur += (size_t)NW;
    int8_t* wqu = (int8_t*)(ws + cur); cur += (size_t)NW;
    int8_t* wqd = (int8_t*)(ws + cur); cur += (size_t)NW;
    float*  hbf = (float*) (ws + cur); cur += (size_t)M * I * sizeof(float);
    int8_t* hq  = (int8_t*)(ws + cur); cur += (size_t)M * I;
    float*  hds = (float*) (ws + cur); cur += (size_t)M * sizeof(float);

    kabs_partial<<<NB, THREADS, 0, stream>>>(wg, NW, partials + 0 * NB);
    kabs_partial<<<NB, THREADS, 0, stream>>>(wu, NW, partials + 1 * NB);
    kabs_partial<<<NB, THREADS, 0, stream>>>(wd, NW, partials + 2 * NB);
    kscale_finalize<<<3, THREADS, 0, stream>>>(partials, scale, descale,
                                               1.f / (float)NW, NB);

    const long long n4 = NW / 4;
    const int qblk = (int)((n4 + THREADS - 1) / THREADS);
    kquant_w<<<qblk, THREADS, 0, stream>>>(wg, wqg, n4, scale, 0);
    kquant_w<<<qblk, THREADS, 0, stream>>>(wu, wqu, n4, scale, 1);
    kquant_w<<<qblk, THREADS, 0, stream>>>(wd, wqd, n4, scale, 2);

    kact_quant<<<M, THREADS, 0, stream>>>(x, xq, xds, D);

    dim3 g1(M / TM, I / TN);
    kgemm_gate_up<<<g1, THREADS, 0, stream>>>(xq, xds, wqg, wqu, descale, hbf, I, D);

    kact_quant<<<M, THREADS, 0, stream>>>(hbf, hq, hds, I);

    dim3 g2(M / TM, D / TN);
    kgemm_down<<<g2, THREADS, 0, stream>>>(hq, hds, wqd, descale, out, D, I);
}
